// BiGRUNer_5068061409863
// MI455X (gfx1250) — matrix operand order repaired and checked by reading, not compiled, not run on hardware
//
#include <hip/hip_runtime.h>
#include <hip/hip_bf16.h>
#include <math.h>

#define BATCHN 32
#define TLEN   256
#define HIDN   768
#define LH     128
#define GHN    384   // 3*LH
#define NL     9

typedef __bf16 bf16_t;
typedef __attribute__((ext_vector_type(16))) __bf16 v16bf;
typedef __attribute__((ext_vector_type(8)))  __bf16 v8bf;
typedef __attribute__((ext_vector_type(8)))  float  v8f;

__device__ __forceinline__ bf16_t f2bf(float f) {
  union { float f; unsigned u; } in; in.f = f;
  unsigned u = in.u;
  unsigned r = (u + 0x7FFFu + ((u >> 16) & 1u)) >> 16;
  union { unsigned short s; bf16_t b; } out; out.s = (unsigned short)r;
  return out.b;
}

__device__ __forceinline__ float sigmoidf(float x) { return 1.0f / (1.0f + __expf(-x)); }

// ---------------------------------------------------------------------------
// 1) Embedding gather + cast to bf16: xbf[(b*T+t)*HID + d] = bf16(emb[ids[b][t]][d])
// ---------------------------------------------------------------------------
__global__ void gather_cast_kernel(const int* __restrict__ ids,
                                   const float* __restrict__ emb,
                                   bf16_t* __restrict__ xbf) {
  long idx = (long)blockIdx.x * blockDim.x + threadIdx.x;
  long total = (long)BATCHN * TLEN * HIDN;
  if (idx >= total) return;
  int d   = (int)(idx % HIDN);
  int row = (int)(idx / HIDN);           // b*T + t
  int tok = ids[row];
  xbf[idx] = f2bf(emb[(long)tok * HIDN + d]);
}

// ---------------------------------------------------------------------------
// 2) Pack Wih (384x768 f32, row-major [g][d]) into WMMA B-fragment layout:
//    pack[((kt*24 + nt)*32 + lane)*16 + e] = bf16(W[nt*16 + lane%16][kt*32 + 16*(lane/16) + e])
// ---------------------------------------------------------------------------
__global__ void pack_wih_kernel(const float* __restrict__ Wf,
                                const float* __restrict__ Wb,
                                bf16_t* __restrict__ pack) {
  const int per_dir = 24 * 24 * 32 * 16;   // 294912
  int idx = blockIdx.x * blockDim.x + threadIdx.x;
  int dir = blockIdx.y;
  if (idx >= per_dir) return;
  int e    = idx & 15;
  int lane = (idx >> 4) & 31;
  int tile = idx >> 9;          // kt*24 + nt
  int nt   = tile % 24;
  int kt   = tile / 24;
  int n = nt * 16 + (lane & 15);
  int k = kt * 32 + 16 * (lane >> 4) + e;
  const float* W = dir ? Wb : Wf;
  pack[(long)dir * per_dir + idx] = f2bf(W[(long)n * HIDN + k]);
}

// ---------------------------------------------------------------------------
// 3) Pack Whh (384x128 f32) the same way (kt in 0..3).
// ---------------------------------------------------------------------------
__global__ void pack_whh_kernel(const float* __restrict__ Wf,
                                const float* __restrict__ Wb,
                                bf16_t* __restrict__ pack) {
  const int per_dir = 4 * 24 * 32 * 16;    // 49152
  int idx = blockIdx.x * blockDim.x + threadIdx.x;
  int dir = blockIdx.y;
  if (idx >= per_dir) return;
  int e    = idx & 15;
  int lane = (idx >> 4) & 31;
  int tile = idx >> 9;          // kt*24 + nt
  int nt   = tile % 24;
  int kt   = tile / 24;
  int n = nt * 16 + (lane & 15);
  int k = kt * 32 + 16 * (lane >> 4) + e;
  const float* W = dir ? Wb : Wf;
  pack[(long)dir * per_dir + idx] = f2bf(W[(long)n * LH + k]);
}

// ---------------------------------------------------------------------------
// 4) Input projection via WMMA bf16:  xp = x @ Wih^T + bih  (8192x768 @ 768x384)
//    Register-blocked 16x96 strip per wave (6 n-tiles share one A fragment)
//    with software-pipelined, double-buffered B fragments: loads for k-step
//    s+1 are issued before the WMMAs of step s, so s_wait_loadcnt overlaps
//    ~6 WMMA issue slots instead of serializing load->wait->wmma.
// ---------------------------------------------------------------------------
#define NTB 6   // n-tiles per wave
__global__ void __launch_bounds__(256)
proj_kernel(const bf16_t* __restrict__ xbf,
            const bf16_t* __restrict__ wih_pack,
            const float* __restrict__ bih_f,
            const float* __restrict__ bih_b,
            float* __restrict__ xp) {
  const int dir  = blockIdx.y;
  const int lane = threadIdx.x & 31;
  const int wid  = threadIdx.x >> 5;
  const int wg   = blockIdx.x * 8 + wid;       // 0..2047
  const int mtile = wg >> 2;                   // 0..511
  const int ngrp  = wg & 3;                    // 0..3 -> ntiles ngrp*6..+5
  const int g     = lane >> 4;                 // lane half
  const int row   = mtile * 16 + (lane & 15);  // A row this lane owns

  const bf16_t* bpackd = wih_pack + (long)dir * (24 * 24 * 512);
  const float*  bih    = dir ? bih_b : bih_f;

  // per-lane base pointer into the packed B (lane-contiguous 32B fragments)
  const bf16_t* bbase = bpackd + ((long)(ngrp * NTB) * 32 + lane) * 16;
  // stride between k-steps in the packed layout: 24 tiles * 512 elems
  const long bkstride = 24 * 512;
  // stride between adjacent n-tiles: 512 elems
  const long bnstride = 512;

  v8f acc[NTB];
  #pragma unroll
  for (int nn = 0; nn < NTB; ++nn) acc[nn] = (v8f){};

  auto loadA = [&](int kt) {
    const bf16_t* rowp = xbf + (long)row * HIDN + kt * 32;
    union { v16bf v16; v8bf v8[2]; } a;
    a.v8[0] = *reinterpret_cast<const v8bf*>(rowp + 16 * g);
    a.v8[1] = *reinterpret_cast<const v8bf*>(rowp + 16 * g + 8);
    return a.v16;
  };

  v16bf b0[NTB], b1[NTB];
  #pragma unroll
  for (int nn = 0; nn < NTB; ++nn)
    b0[nn] = *reinterpret_cast<const v16bf*>(bbase + nn * bnstride);

  for (int kt = 0; kt < 24; kt += 2) {
    // issue loads for step kt+1 before consuming step kt
    v16bf a0 = loadA(kt);
    #pragma unroll
    for (int nn = 0; nn < NTB; ++nn)
      b1[nn] = *reinterpret_cast<const v16bf*>(
          bbase + (long)(kt + 1) * bkstride + nn * bnstride);
    #pragma unroll
    for (int nn = 0; nn < NTB; ++nn)
      acc[nn] = __builtin_amdgcn_wmma_f32_16x16x32_bf16(false, a0, false, b0[nn],
                                                        (short)0, acc[nn], false, false);
    // issue loads for step kt+2 before consuming step kt+1
    v16bf a1 = loadA(kt + 1);
    if (kt + 2 < 24) {
      #pragma unroll
      for (int nn = 0; nn < NTB; ++nn)
        b0[nn] = *reinterpret_cast<const v16bf*>(
            bbase + (long)(kt + 2) * bkstride + nn * bnstride);
    }
    #pragma unroll
    for (int nn = 0; nn < NTB; ++nn)
      acc[nn] = __builtin_amdgcn_wmma_f32_16x16x32_bf16(false, a1, false, b1[nn],
                                                        (short)0, acc[nn], false, false);
  }

  // Scatter C: elem e -> (row mtile*16 + e + 8*g, col nt*16 + lane%16)
  #pragma unroll
  for (int nn = 0; nn < NTB; ++nn) {
    const int n = (ngrp * NTB + nn) * 16 + (lane & 15);
    const float bias = bih[n];
    #pragma unroll
    for (int e = 0; e < 8; ++e) {
      int r  = mtile * 16 + e + 8 * g;   // b*T + t
      int bb = r >> 8;
      int t  = r & 255;
      int tp = dir ? (255 - t) : t;
      xp[(long)dir * (TLEN * BATCHN * GHN) + ((long)tp * BATCHN + bb) * GHN + n] =
          acc[nn][e] + bias;
    }
  }
}

// ---------------------------------------------------------------------------
// 5) GRU recurrent scan, one workgroup per direction (grid.x = 2), 8 waves.
//    h (32x128) lives in LDS; gh = h @ Whh^T via WMMA. Whh B-fragments are
//    hoisted into registers (12 x v16bf per wave) before the 256-step loop,
//    so the serial inner loop touches no global memory for weights.
// ---------------------------------------------------------------------------
__global__ void __launch_bounds__(256)
gru_scan_kernel(const float* __restrict__ xp_all,
                const bf16_t* __restrict__ whh_pack,
                const float* __restrict__ bhh_f,
                const float* __restrict__ bhh_b,
                float* __restrict__ hseq_all) {
  __shared__ float h_lds[BATCHN * LH];    // 16 KB
  __shared__ float gh_lds[BATCHN * GHN];  // 48 KB

  const int dir  = blockIdx.x;
  const int tid  = threadIdx.x;
  const int lane = tid & 31;
  const int wid  = tid >> 5;
  const int g    = lane >> 4;

  const float*  xp    = xp_all   + (long)dir * (TLEN * BATCHN * GHN);
  const bf16_t* bpack = whh_pack + (long)dir * (4 * 24 * 512);
  const float*  bhh   = dir ? bhh_b : bhh_f;
  float*        hseq  = hseq_all + (long)dir * (TLEN * BATCHN * LH);

  const int nt0 = wid * 3;  // this wave owns ntiles nt0..nt0+2, both mtiles

  // Hoist all Whh B fragments for this wave into registers (3 nt x 4 kt).
  v16bf bfrag[3][4];
  #pragma unroll
  for (int nn = 0; nn < 3; ++nn)
    #pragma unroll
    for (int kt = 0; kt < 4; ++kt)
      bfrag[nn][kt] = *reinterpret_cast<const v16bf*>(
          bpack + ((long)(kt * 24 + (nt0 + nn)) * 32 + lane) * 16);

  // h0 = 0
  for (int i = tid; i < BATCHN * LH; i += 256) h_lds[i] = 0.0f;
  __syncthreads();

  for (int t = 0; t < TLEN; ++t) {
    // ---- gh = h @ Whh^T (WMMA) ----
    #pragma unroll
    for (int mt = 0; mt < 2; ++mt) {
      const int m = mt * 16 + (lane & 15);
      // A fragments for all 4 k-tiles from LDS h (f32 -> bf16)
      v16bf afrag[4];
      #pragma unroll
      for (int kt = 0; kt < 4; ++kt) {
        v16bf a;
        const int kbase = kt * 32 + 16 * g;
        #pragma unroll
        for (int e = 0; e < 8; ++e)  a[e]     = f2bf(h_lds[m * LH + kbase + e]);
        #pragma unroll
        for (int e = 0; e < 8; ++e)  a[8 + e] = f2bf(h_lds[m * LH + kbase + 8 + e]);
        afrag[kt] = a;
      }
      #pragma unroll
      for (int nn = 0; nn < 3; ++nn) {
        v8f acc = {};
        #pragma unroll
        for (int kt = 0; kt < 4; ++kt)
          acc = __builtin_amdgcn_wmma_f32_16x16x32_bf16(false, afrag[kt], false,
                                                        bfrag[nn][kt],
                                                        (short)0, acc, false, false);
        const int n = (nt0 + nn) * 16 + (lane & 15);
        #pragma unroll
        for (int e = 0; e < 8; ++e)
          gh_lds[(mt * 16 + e + 8 * g) * GHN + n] = acc[e];
      }
    }
    __syncthreads();

    // ---- fused gate update: 4096 elements over 256 threads ----
    {
      const int b     = tid >> 3;
      const int jbase = (tid & 7) * 16;
      const long xrow = ((long)t * BATCHN + b) * GHN;
      const int  tout = dir ? (255 - t) : t;
      #pragma unroll
      for (int i = 0; i < 16; ++i) {
        const int j = jbase + i;
        const float hprev = h_lds[b * LH + j];
        const float r = sigmoidf(xp[xrow + j]       + gh_lds[b * GHN + j]       + bhh[j]);
        const float z = sigmoidf(xp[xrow + 128 + j] + gh_lds[b * GHN + 128 + j] + bhh[128 + j]);
        const float n = tanhf(xp[xrow + 256 + j] +
                              r * (gh_lds[b * GHN + 256 + j] + bhh[256 + j]));
        const float hn = (1.0f - z) * n + z * hprev;
        h_lds[b * LH + j] = hn;
        hseq[((long)tout * BATCHN + b) * LH + j] = hn;
      }
    }
    __syncthreads();
  }
}

// ---------------------------------------------------------------------------
// 6) Emissions: e[b][t][c] = blin[c] + hf.Wlin[c,0:128] + hb.Wlin[c,128:256]
// ---------------------------------------------------------------------------
__global__ void emis_kernel(const float* __restrict__ hf,
                            const float* __restrict__ hb,
                            const float* __restrict__ Wlin,
                            const float* __restrict__ blin,
                            float* __restrict__ emis) {
  int idx = blockIdx.x * blockDim.x + threadIdx.x;
  if (idx >= BATCHN * TLEN * NL) return;
  int c  = idx % NL;
  int bt = idx / NL;
  int b  = bt / TLEN;
  int t  = bt % TLEN;
  const float* hfp = hf + ((long)t * BATCHN + b) * LH;
  const float* hbp = hb + ((long)t * BATCHN + b) * LH;
  const float* w   = Wlin + (long)c * (2 * LH);
  float acc = blin[c];
  for (int j = 0; j < LH; ++j) acc += hfp[j] * w[j];
  for (int j = 0; j < LH; ++j) acc += hbp[j] * w[LH + j];
  emis[((long)b * TLEN + t) * NL + c] = acc;
}

// ---------------------------------------------------------------------------
// 7) Viterbi decode: one thread per batch row; backpointers in ws.
// ---------------------------------------------------------------------------
__global__ void viterbi_kernel(const float* __restrict__ emis,
                               const float* __restrict__ trans,
                               const float* __restrict__ start,
                               const float* __restrict__ endv,
                               int* __restrict__ bp,
                               float* __restrict__ out) {
  __shared__ float tr[NL * NL], st[NL], en[NL];
  int b = threadIdx.x;
  if (b < NL * NL) tr[b] = trans[b];
  if (b < NL) { st[b] = start[b]; en[b] = endv[b]; }
  __syncthreads();
  if (b >= BATCHN) return;

  float alpha[NL], anew[NL];
  for (int j = 0; j < NL; ++j) alpha[j] = st[j] + emis[((long)b * TLEN) * NL + j];
  for (int t = 1; t < TLEN; ++t) {
    for (int j = 0; j < NL; ++j) {
      float best = alpha[0] + tr[j];
      int bi = 0;
      for (int i = 1; i < NL; ++i) {
        float s = alpha[i] + tr[i * NL + j];
        if (s > best) { best = s; bi = i; }
      }
      anew[j] = best + emis[((long)b * TLEN + t) * NL + j];
      bp[((long)(t - 1) * BATCHN + b) * NL + j] = bi;
    }
    for (int j = 0; j < NL; ++j) alpha[j] = anew[j];
  }
  float best = alpha[0] + en[0];
  int y = 0;
  for (int j = 1; j < NL; ++j) {
    float s = alpha[j] + en[j];
    if (s > best) { best = s; y = j; }
  }
  out[(long)b * TLEN + 255] = (float)y;
  for (int t = TLEN - 2; t >= 0; --t) {
    y = bp[((long)t * BATCHN + b) * NL + y];
    out[(long)b * TLEN + t] = (float)y;
  }
}

// ---------------------------------------------------------------------------
// 8) CRF negative mean log-likelihood -> out[B*T]
// ---------------------------------------------------------------------------
__global__ void crf_kernel(const float* __restrict__ emis,
                           const int* __restrict__ labels,
                           const float* __restrict__ trans,
                           const float* __restrict__ start,
                           const float* __restrict__ endv,
                           float* __restrict__ out) {
  __shared__ float tr[NL * NL], st[NL], en[NL], red[BATCHN];
  int b = threadIdx.x;
  if (b < NL * NL) tr[b] = trans[b];
  if (b < NL) { st[b] = start[b]; en[b] = endv[b]; }
  __syncthreads();
  if (b < BATCHN) {
    const int* lab = labels + (long)b * TLEN;
    float num = st[lab[0]];
    for (int t = 0; t < TLEN; ++t) num += emis[((long)b * TLEN + t) * NL + lab[t]];
    for (int t = 1; t < TLEN; ++t) num += tr[lab[t - 1] * NL + lab[t]];
    num += en[lab[TLEN - 1]];

    float alpha[NL], anew[NL];
    for (int j = 0; j < NL; ++j) alpha[j] = st[j] + emis[((long)b * TLEN) * NL + j];
    for (int t = 1; t < TLEN; ++t) {
      for (int j = 0; j < NL; ++j) {
        float m = alpha[0] + tr[j];
        for (int i = 1; i < NL; ++i) {
          float s = alpha[i] + tr[i * NL + j];
          m = fmaxf(m, s);
        }
        float sum = 0.0f;
        for (int i = 0; i < NL; ++i) sum += __expf(alpha[i] + tr[i * NL + j] - m);
        anew[j] = m + __logf(sum) + emis[((long)b * TLEN + t) * NL + j];
      }
      for (int j = 0; j < NL; ++j) alpha[j] = anew[j];
    }
    float m = alpha[0] + en[0];
    for (int j = 1; j < NL; ++j) m = fmaxf(m, alpha[j] + en[j]);
    float sum = 0.0f;
    for (int j = 0; j < NL; ++j) sum += __expf(alpha[j] + en[j] - m);
    float denom = m + __logf(sum);
    red[b] = num - denom;
  }
  __syncthreads();
  if (b == 0) {
    float s = 0.0f;
    for (int i = 0; i < BATCHN; ++i) s += red[i];
    out[(long)BATCHN * TLEN] = -s / (float)BATCHN;
  }
}

// ---------------------------------------------------------------------------
extern "C" void kernel_launch(void* const* d_in, const int* in_sizes, int n_in,
                              void* d_out, int out_size, void* d_ws, size_t ws_size,
                              hipStream_t stream) {
  const int*   ids   = (const int*)d_in[0];
  const int*   labs  = (const int*)d_in[2];
  const float* emb   = (const float*)d_in[3];
  const float* Wih_f = (const float*)d_in[4];
  const float* Whh_f = (const float*)d_in[5];
  const float* bih_f = (const float*)d_in[6];
  const float* bhh_f = (const float*)d_in[7];
  const float* Wih_b = (const float*)d_in[8];
  const float* Whh_b = (const float*)d_in[9];
  const float* bih_b = (const float*)d_in[10];
  const float* bhh_b = (const float*)d_in[11];
  const float* Wlin  = (const float*)d_in[12];
  const float* blin  = (const float*)d_in[13];
  const float* trans = (const float*)d_in[14];
  const float* start = (const float*)d_in[15];
  const float* endv  = (const float*)d_in[16];
  float* out = (float*)d_out;

  // workspace carving (256B aligned)
  size_t off = 0;
  auto carve = [&](size_t bytes) -> char* {
    char* p = (char*)d_ws + off;
    off = (off + bytes + 255) & ~(size_t)255;
    return p;
  };
  bf16_t* xbf      = (bf16_t*)carve((size_t)BATCHN * TLEN * HIDN * 2);       // 12.6 MB
  bf16_t* wih_pack = (bf16_t*)carve((size_t)2 * 24 * 24 * 512 * 2);          // 1.2 MB
  bf16_t* whh_pack = (bf16_t*)carve((size_t)2 * 4 * 24 * 512 * 2);           // 0.2 MB
  float*  xp       = (float*)carve((size_t)2 * TLEN * BATCHN * GHN * 4);     // 25.2 MB
  float*  hseq     = (float*)carve((size_t)2 * TLEN * BATCHN * LH * 4);      // 8.4 MB
  float*  emis     = (float*)carve((size_t)BATCHN * TLEN * NL * 4);          // 0.3 MB
  int*    bp       = (int*)carve((size_t)(TLEN - 1) * BATCHN * NL * 4);      // 0.3 MB
  (void)ws_size;

  // 1) gather + bf16 cast
  {
    long total = (long)BATCHN * TLEN * HIDN;
    gather_cast_kernel<<<dim3((unsigned)((total + 255) / 256)), dim3(256), 0, stream>>>(
        ids, emb, xbf);
  }
  // 2/3) pack weights into WMMA B-fragment layout
  pack_wih_kernel<<<dim3((24 * 24 * 512 + 255) / 256, 2), dim3(256), 0, stream>>>(
      Wih_f, Wih_b, wih_pack);
  pack_whh_kernel<<<dim3((4 * 24 * 512 + 255) / 256, 2), dim3(256), 0, stream>>>(
      Whh_f, Whh_b, whh_pack);
  // 4) input projection (WMMA), 16x96 strip per wave, double-buffered B
  proj_kernel<<<dim3(256, 2), dim3(256), 0, stream>>>(xbf, wih_pack, bih_f, bih_b, xp);
  // 5) sequential GRU scan (WMMA), one workgroup per direction
  gru_scan_kernel<<<dim3(2), dim3(256), 0, stream>>>(xp, whh_pack, bhh_f, bhh_b, hseq);
  // 6) emissions
  emis_kernel<<<dim3((BATCHN * TLEN * NL + 255) / 256), dim3(256), 0, stream>>>(
      hseq, hseq + (size_t)TLEN * BATCHN * LH, Wlin, blin, emis);
  // 7) viterbi decode -> out[0 .. B*T)
  viterbi_kernel<<<dim3(1), dim3(BATCHN * NL), 0, stream>>>(emis, trans, start, endv, bp, out);
  // 8) CRF loss -> out[B*T]
  crf_kernel<<<dim3(1), dim3(BATCHN * NL), 0, stream>>>(emis, labs, trans, start, endv, out);
}
